// HLmEncGenLayer_9354438771313
// MI455X (gfx1250) — matrix operand order repaired and checked by reading, not compiled, not run on hardware
//
#include <hip/hip_runtime.h>
#include <math.h>

#define THREADS 256

typedef __attribute__((ext_vector_type(16))) __bf16 v16bf;
typedef __attribute__((ext_vector_type(8)))  float  v8f;

struct __align__(16) U4 { unsigned x, y, z, w; };

union AF { v16bf v; U4 q[2]; };

// native bf16 convert (hardware cvt, round-to-nearest-even)
__device__ __forceinline__ unsigned short f2bf(float f) {
  union { __bf16 b; unsigned short s; } u;
  u.b = (__bf16)f;
  return u.s;
}

__device__ __forceinline__ v8f wmma_bf16(v16bf a, v16bf b, v8f c) {
  return __builtin_amdgcn_wmma_f32_16x16x32_bf16(false, a, false, b, (short)0, c, false, false);
}

// ---- CDNA5 async global->LDS copy (ASYNCcnt path), with safe fallback ----
#if defined(__gfx1250__) && __has_builtin(__builtin_amdgcn_global_load_async_to_lds_b128)
#define ASYNC_COPY 1
typedef int v4i __attribute__((vector_size(16)));
typedef __attribute__((address_space(1))) v4i v4i_g;   // global
typedef __attribute__((address_space(3))) v4i v4i_l;   // LDS
#endif

__device__ __forceinline__ void copy16(const unsigned short* g, unsigned short* l) {
#if ASYNC_COPY
  __builtin_amdgcn_global_load_async_to_lds_b128((v4i_g*)g, (v4i_l*)l, 0, 0);
#else
  *(U4*)l = *(const U4*)g;
#endif
}

__device__ __forceinline__ void zero16(unsigned short* l) {
  U4 z; z.x = 0; z.y = 0; z.z = 0; z.w = 0;
  *(U4*)l = z;
}

__device__ __forceinline__ void async_wait0() {
#if ASYNC_COPY
#if __has_builtin(__builtin_amdgcn_s_wait_asynccnt)
  __builtin_amdgcn_s_wait_asynccnt(0);
#else
  asm volatile("s_wait_asynccnt 0x0" ::: "memory");
#endif
#endif
}

// ---------------------------------------------------------------------------
// WMMA GEMM: C[M,N] = A[M,K](lda, bf16) * B[N,K](bf16)^T (+ bias[N]), f32 acc.
// 128x128x32 tiles, 8 waves. Double-buffered LDS staging with async b128
// copies overlapped against the WMMA stream. Requires M % 128 == 0 (true for
// all call sites: M = 2048).
// ---------------------------------------------------------------------------
__global__ __launch_bounds__(THREADS)
void k_gemm(const unsigned short* __restrict__ A, int lda,
            const unsigned short* __restrict__ B,
            const float* __restrict__ bias,
            float* __restrict__ C, int M, int N, int K)
{
  __shared__ unsigned short As[2][128 * 40];
  __shared__ unsigned short Bs[2][128 * 40];
  const int tid  = threadIdx.x;
  const int lane = tid & 31, wid = tid >> 5;
  const int hf = lane >> 4, l16 = lane & 15;
  const int wm = wid & 3, wn = wid >> 2;
  const int rowBlk = blockIdx.y * 128, colBlk = blockIdx.x * 128;

  // the two 16B chunks this thread stages per 128x32 tile (K-invariant coords)
  const int r0 = tid >> 2,         c0 = (tid & 3) * 8;
  const int r1 = r0 + 64,          c1 = c0;
  const bool bv0 = (colBlk + r0) < N;
  const bool bv1 = (colBlk + r1) < N;

  // zero-fill out-of-range B rows once; they stay zero across all K steps
  if (colBlk + 128 > N) {
    if (!bv0) { zero16(&Bs[0][r0 * 40 + c0]); zero16(&Bs[1][r0 * 40 + c0]); }
    if (!bv1) { zero16(&Bs[0][r1 * 40 + c1]); zero16(&Bs[1][r1 * 40 + c1]); }
  }

  v8f acc[2][4];
#pragma unroll
  for (int i = 0; i < 2; ++i)
#pragma unroll
    for (int j = 0; j < 4; ++j) acc[i][j] = {};

  // prologue: stage K-step 0 into buffer 0
  copy16(A + (size_t)(rowBlk + r0) * lda + c0, &As[0][r0 * 40 + c0]);
  copy16(A + (size_t)(rowBlk + r1) * lda + c1, &As[0][r1 * 40 + c1]);
  if (bv0) copy16(B + (size_t)(colBlk + r0) * K + c0, &Bs[0][r0 * 40 + c0]);
  if (bv1) copy16(B + (size_t)(colBlk + r1) * K + c1, &Bs[0][r1 * 40 + c1]);
  async_wait0();
  __syncthreads();

  int buf = 0;
  for (int k0 = 0; k0 < K; k0 += 32) {
    // prefetch next K-step into the other buffer while we compute this one
    const int k1 = k0 + 32;
    if (k1 < K) {
      unsigned short* An = As[buf ^ 1];
      unsigned short* Bn = Bs[buf ^ 1];
      copy16(A + (size_t)(rowBlk + r0) * lda + k1 + c0, An + r0 * 40 + c0);
      copy16(A + (size_t)(rowBlk + r1) * lda + k1 + c1, An + r1 * 40 + c1);
      if (bv0) copy16(B + (size_t)(colBlk + r0) * K + k1 + c0, Bn + r0 * 40 + c0);
      if (bv1) copy16(B + (size_t)(colBlk + r1) * K + k1 + c1, Bn + r1 * 40 + c1);
    }

    const unsigned short* Ac = As[buf];
    const unsigned short* Bc = Bs[buf];
    AF af[2], bfm[4];
#pragma unroll
    for (int i = 0; i < 2; ++i) {
      int r = wm * 32 + i * 16 + l16;
      af[i].q[0] = *(const U4*)&Ac[r * 40 + hf * 8];
      af[i].q[1] = *(const U4*)&Ac[r * 40 + hf * 8 + 16];
    }
#pragma unroll
    for (int j = 0; j < 4; ++j) {
      int n = wn * 64 + j * 16 + l16;
      bfm[j].q[0] = *(const U4*)&Bc[n * 40 + hf * 8];
      bfm[j].q[1] = *(const U4*)&Bc[n * 40 + hf * 8 + 16];
    }
#pragma unroll
    for (int i = 0; i < 2; ++i)
#pragma unroll
      for (int j = 0; j < 4; ++j)
        acc[i][j] = wmma_bf16(af[i].v, bfm[j].v, acc[i][j]);

    async_wait0();
    __syncthreads();
    buf ^= 1;
  }

#pragma unroll
  for (int i = 0; i < 2; ++i)
#pragma unroll
    for (int j = 0; j < 4; ++j)
#pragma unroll
      for (int jj = 0; jj < 8; ++jj) {
        int gr = rowBlk + wm * 32 + i * 16 + jj + hf * 8;
        int gc = colBlk + wn * 64 + j * 16 + l16;
        if (gr < M && gc < N) {
          float val = acc[i][j][jj];
          if (bias) val += bias[gc];
          C[(size_t)gr * N + gc] = val;
        }
      }
}

// ---------------------------------------------------------------------------
// Flash attention, WMMA, online softmax. bh = (batch4, head16). keys padded
// to 1056 (zeros); mask input has 1032 valid key columns. Output bf16.
// ---------------------------------------------------------------------------
__global__ __launch_bounds__(THREADS)
void k_attn(const unsigned short* __restrict__ Q,
            const unsigned short* __restrict__ Kb,
            const unsigned short* __restrict__ Vt,
            const float* __restrict__ emask,
            const float* __restrict__ gmask,
            unsigned short* __restrict__ Out)
{
  __shared__ unsigned short Pl[8][16][32];
  const int bh = blockIdx.x;
  const int b = bh >> 4, h = bh & 15;
  const int wid = threadIdx.x >> 5;
  const int lane = threadIdx.x & 31;
  const int hf = lane >> 4, l16 = lane & 15;
  const int q0 = blockIdx.y * 128 + wid * 16;

  const float* am = (b < 2)
      ? (emask + (size_t)(b * 16 + h) * 1024 * 1032)
      : (gmask + (size_t)((b - 2) * 16 + h) * 1024 * 1032);

  AF qf0, qf1;
  {
    const unsigned short* qp = Q + ((size_t)bh * 1024 + q0 + l16) * 64;
    const int kb = hf * 8;
    qf0.q[0] = *(const U4*)(qp + kb);
    qf0.q[1] = *(const U4*)(qp + kb + 16);
    qf1.q[0] = *(const U4*)(qp + 32 + kb);
    qf1.q[1] = *(const U4*)(qp + 32 + kb + 16);
  }

  v8f oa[4];
#pragma unroll
  for (int t = 0; t < 4; ++t) oa[t] = {};
  float mst[8], lst[8];
#pragma unroll
  for (int j = 0; j < 8; ++j) { mst[j] = -3.0e38f; lst[j] = 0.f; }

  for (int kk = 0; kk < 1056; kk += 32) {
    float a0[8], a1[8], rm0[8], rm1[8];
#pragma unroll
    for (int sub = 0; sub < 2; ++sub) {
      const int key0 = kk + sub * 16;
      AF kf0, kf1;
      const unsigned short* kp = Kb + ((size_t)bh * 1056 + key0 + l16) * 64;
      const int kb2 = hf * 8;
      kf0.q[0] = *(const U4*)(kp + kb2);
      kf0.q[1] = *(const U4*)(kp + kb2 + 16);
      kf1.q[0] = *(const U4*)(kp + 32 + kb2);
      kf1.q[1] = *(const U4*)(kp + 32 + kb2 + 16);
      v8f s = {};
      s = wmma_bf16(qf0.v, kf0.v, s);
      s = wmma_bf16(qf1.v, kf1.v, s);
      const int key = key0 + l16;
#pragma unroll
      for (int j = 0; j < 8; ++j) {
        const int r = j + hf * 8;
        float val = s[j] * 0.125f;  // 1/sqrt(64)
        val += (key < 1032) ? am[(size_t)(q0 + r) * 1032 + key] : -1.0e9f;
        float vm = val;
        vm = fmaxf(vm, __shfl_xor(vm, 1, 32));
        vm = fmaxf(vm, __shfl_xor(vm, 2, 32));
        vm = fmaxf(vm, __shfl_xor(vm, 4, 32));
        vm = fmaxf(vm, __shfl_xor(vm, 8, 32));
        if (sub == 0) { a0[j] = val; rm0[j] = vm; }
        else          { a1[j] = val; rm1[j] = vm; }
      }
    }
    float sc_[8];
#pragma unroll
    for (int j = 0; j < 8; ++j) {
      const int r = j + hf * 8;
      float nm = fmaxf(mst[j], fmaxf(rm0[j], rm1[j]));
      float scl = __expf(mst[j] - nm);
      float p0 = __expf(a0[j] - nm);
      float p1 = __expf(a1[j] - nm);
      float rs = p0 + p1;
      rs += __shfl_xor(rs, 1, 32);
      rs += __shfl_xor(rs, 2, 32);
      rs += __shfl_xor(rs, 4, 32);
      rs += __shfl_xor(rs, 8, 32);
      lst[j] = lst[j] * scl + rs;
      mst[j] = nm;
      sc_[j] = scl;
      Pl[wid][r][l16]      = f2bf(p0);
      Pl[wid][r][16 + l16] = f2bf(p1);
    }
#pragma unroll
    for (int t = 0; t < 4; ++t)
#pragma unroll
      for (int j = 0; j < 8; ++j) oa[t][j] *= sc_[j];

    AF pf;
    {
      const unsigned short* pp = &Pl[wid][l16][0];
      const int kb = hf * 8;
      pf.q[0] = *(const U4*)(pp + kb);
      pf.q[1] = *(const U4*)(pp + kb + 16);
    }
#pragma unroll
    for (int t = 0; t < 4; ++t) {
      AF vf;
      const unsigned short* vp = Vt + ((size_t)bh * 64 + t * 16 + l16) * 1056 + kk;
      const int kb2 = hf * 8;
      vf.q[0] = *(const U4*)(vp + kb2);
      vf.q[1] = *(const U4*)(vp + kb2 + 16);
      oa[t] = wmma_bf16(pf.v, vf.v, oa[t]);
    }
  }

#pragma unroll
  for (int j = 0; j < 8; ++j) {
    const int r = j + hf * 8;
    float inv = 1.f / lst[j];
    size_t base = ((size_t)(b * 1024) + q0 + r) * 1024 + (size_t)h * 64;
#pragma unroll
    for (int t = 0; t < 4; ++t) Out[base + t * 16 + l16] = f2bf(oa[t][j] * inv);
  }
}

// ---------------------------------------------------------------------------
// Elementwise / prep kernels
// ---------------------------------------------------------------------------
__global__ void k_f2bfv(const float* __restrict__ src, unsigned short* __restrict__ dst, int n) {
  int i = blockIdx.x * THREADS + threadIdx.x;
  if (i < n) dst[i] = f2bf(src[i]);
}

__global__ void k_mask_wqkv(const float* __restrict__ Wsrc, unsigned short* __restrict__ Wdst, int n) {
  int i = blockIdx.x * THREADS + threadIdx.x;
  if (i >= n) return;
  int r = i / 1056, c = i - r * 1056;
  float w = Wsrc[i];
  if (c >= 1024) {
    int rr = r & 1023;
    if (!(r >= 1024 && rr < 256)) w = 0.f;   // QKV_MASK: noise feeds only first-NIAF k/v rows
  }
  Wdst[i] = f2bf(w);
}

__global__ void k_mask_noise(const float* __restrict__ noise, const int* __restrict__ mask,
                             float* __restrict__ nz, int n) {
  int i = blockIdx.x * THREADS + threadIdx.x;
  if (i < n) nz[i] = noise[i] * (float)mask[i >> 5];
}

__global__ void k_fill_noise(const float* __restrict__ nz, unsigned short* __restrict__ xae, int n) {
  int i = blockIdx.x * THREADS + threadIdx.x;
  if (i >= n) return;
  int t = i >> 5, zi = i & 31;
  xae[(size_t)t * 1056 + 1024 + zi] = f2bf(nz[i]);
}

__global__ __launch_bounds__(THREADS)
void k_ln(const float* __restrict__ x, const int* __restrict__ mask,
          const float* __restrict__ sc, const float* __restrict__ bi,
          unsigned short* __restrict__ out, int ostride)
{
  __shared__ float red[THREADS];
  const int t = blockIdx.x;
  const float* row = x + (size_t)t * 1024;
  float s = 0.f;
  for (int i = threadIdx.x; i < 1024; i += THREADS) s += row[i];
  red[threadIdx.x] = s; __syncthreads();
  for (int o = THREADS / 2; o > 0; o >>= 1) {
    if (threadIdx.x < o) red[threadIdx.x] += red[threadIdx.x + o];
    __syncthreads();
  }
  float mean = red[0] * (1.f / 1024.f);
  __syncthreads();
  float v = 0.f;
  for (int i = threadIdx.x; i < 1024; i += THREADS) { float d = row[i] - mean; v += d * d; }
  red[threadIdx.x] = v; __syncthreads();
  for (int o = THREADS / 2; o > 0; o >>= 1) {
    if (threadIdx.x < o) red[threadIdx.x] += red[threadIdx.x + o];
    __syncthreads();
  }
  float rstd = rsqrtf(red[0] * (1.f / 1024.f) + 1e-5f);
  int m = (sc && mask) ? mask[t] : 0;
  for (int i = threadIdx.x; i < 1024; i += THREADS) {
    float nv = (row[i] - mean) * rstd;
    if (sc) nv = bi[m * 1024 + i] + (1.f + sc[m * 1024 + i]) * nv;
    out[(size_t)t * ostride + i] = f2bf(nv);
  }
}

__global__ void k_build_q(const float* __restrict__ qkve, const float* __restrict__ qkvg,
                          unsigned short* __restrict__ qb, int n) {
  int idx = blockIdx.x * THREADS + threadIdx.x;
  if (idx >= n) return;
  int bidx = idx >> 20;
  int rem = idx & ((1 << 20) - 1);
  int s = rem >> 10, c = rem & 1023;
  int h = c >> 6, d = c & 63;
  const float* src = (bidx < 2 ? qkve : qkvg) + (size_t)((bidx & 1) * 1024 + s) * 3072;
  float x  = src[c];
  float x2 = src[(c & ~63) | (d ^ 32)];
  int i32 = d & 31;
  float ang = (float)s * __expf(-(float)i32 * (9.210340371976184f / 32.f));
  float cs = __cosf(ang), sn = __sinf(ang);
  float r = (d < 32) ? (x * cs - x2 * sn) : (x * cs + x2 * sn);
  qb[((size_t)(bidx * 16 + h) * 1024 + s) * 64 + d] = f2bf(r);
}

__global__ void k_build_kv(const float* __restrict__ qkve, const float* __restrict__ qkvg,
                           const float* __restrict__ eregs, const float* __restrict__ gregs,
                           unsigned short* __restrict__ Kb, unsigned short* __restrict__ Vt, int n) {
  int idx = blockIdx.x * THREADS + threadIdx.x;
  if (idx >= n) return;
  int bidx = idx / (1056 * 1024);
  int rem = idx - bidx * (1056 * 1024);
  int key = rem >> 10, c = rem & 1023;
  int h = c >> 6, d = c & 63;
  float kv = 0.f, vv = 0.f;
  if (key < 1024) {
    const float* src = (bidx < 2 ? qkve : qkvg) + (size_t)((bidx & 1) * 1024 + key) * 3072;
    float kx = src[1024 + c];
    float k2 = src[1024 + ((c & ~63) | (d ^ 32))];
    vv = src[2048 + c];
    int i32 = d & 31;
    float ang = (float)key * __expf(-(float)i32 * (9.210340371976184f / 32.f));
    float cs = __cosf(ang), sn = __sinf(ang);
    kv = (d < 32) ? (kx * cs - k2 * sn) : (kx * cs + k2 * sn);
  } else if (key < 1032) {
    const float* regs = (bidx < 2 ? eregs : gregs) + (size_t)(key - 1024) * 2048;
    kv = regs[c];          // registers: RoPE at pos 0 == identity
    vv = regs[1024 + c];
  }
  Kb[((size_t)(bidx * 16 + h) * 1056 + key) * 64 + d] = f2bf(kv);
  Vt[((size_t)(bidx * 16 + h) * 64 + d) * 1056 + key] = f2bf(vv);
}

__global__ void k_glu(const float* __restrict__ g, unsigned short* __restrict__ out,
                      int halfN, int n) {
  int i = blockIdx.x * THREADS + threadIdx.x;
  if (i >= n) return;
  int t = i / halfN, j = i - t * halfN;
  float a = g[(size_t)t * 2 * halfN + j];
  float b = g[(size_t)t * 2 * halfN + halfN + j];
  out[i] = f2bf((a / (1.f + __expf(-a))) * b);
}

__global__ void k_res_alpha(float* __restrict__ out, const float* __restrict__ in,
                            const float* __restrict__ delta, const float* __restrict__ alpha, int n) {
  int i = blockIdx.x * THREADS + threadIdx.x;
  if (i < n) out[i] = in[i] + alpha[0] * delta[i];
}

__global__ void k_res_filter(float* __restrict__ out, const float* __restrict__ in,
                             const float* __restrict__ delta, const float* __restrict__ filt,
                             const int* __restrict__ mask, int n) {
  int i = blockIdx.x * THREADS + threadIdx.x;
  if (i >= n) return;
  int t = i >> 10, c = i & 1023;
  out[i] = in[i] + filt[mask[t] * 1024 + c] * delta[i];
}

__global__ void k_zfin(const float* __restrict__ zo, const float* __restrict__ nz,
                       const int* __restrict__ mask, unsigned short* __restrict__ zbuf,
                       float* __restrict__ oz, float* __restrict__ omu,
                       float* __restrict__ osg, int n) {
  int i = blockIdx.x * THREADS + threadIdx.x;
  if (i >= n) return;
  int t = i >> 5, zi = i & 31;
  float fm = (float)mask[t];
  const float ZS = 0.04419417382415922f;              // sqrt(1/(Z*NLAYERS))
  float mu = zo[(size_t)t * 64 + zi] * fm * ZS;
  float ls = zo[(size_t)t * 64 + 32 + zi] * fm * ZS;
  float sp = ls + 0.5413248546129181f;                // log(e-1)
  float sig = (sp > 20.f) ? sp : logf(1.f + __expf(sp));
  float z = mu + sig * nz[i];
  zbuf[i] = f2bf(z); oz[i] = z; omu[i] = mu; osg[i] = sig;
}

__global__ void k_genmu(const float* __restrict__ gin, const int* __restrict__ mask,
                        unsigned short* __restrict__ gmb, float* __restrict__ ogm, int n) {
  int i = blockIdx.x * THREADS + threadIdx.x;
  if (i >= n) return;
  int t = i >> 5;
  float g = gin[i] * (float)mask[t] * 0.04419417382415922f;
  gmb[i] = f2bf(g); ogm[i] = g;
}

// ---------------------------------------------------------------------------
// Host orchestration
// ---------------------------------------------------------------------------
extern "C" void kernel_launch(void* const* d_in, const int* in_sizes, int n_in,
                              void* d_out, int out_size, void* d_ws, size_t ws_size,
                              hipStream_t stream)
{
  (void)in_sizes; (void)n_in; (void)out_size; (void)ws_size;
  const float* enc_states = (const float*)d_in[0];
  const float* gen_states = (const float*)d_in[1];
  const float* noise      = (const float*)d_in[2];
  const int*   mask       = (const int*)d_in[3];
  const float* emask      = (const float*)d_in[4];
  const float* gmask      = (const float*)d_in[5];
  const float* alpha_attn = (const float*)d_in[6];
  const float* alpha_mlp  = (const float*)d_in[7];
  const float* alpha_z    = (const float*)d_in[8];
  const float* gas  = (const float*)d_in[9];
  const float* gab  = (const float*)d_in[10];
  const float* gaf  = (const float*)d_in[11];
  const float* gms  = (const float*)d_in[12];
  const float* gmbp = (const float*)d_in[13];
  const float* gmf  = (const float*)d_in[14];
  const float* gzs  = (const float*)d_in[15];
  const float* gzb  = (const float*)d_in[16];
  const float* gzf  = (const float*)d_in[17];
  const float* W_eau = (const float*)d_in[18];
  const float* b_eau = (const float*)d_in[19];
  const float* W_gau = (const float*)d_in[20];
  const float* b_gau = (const float*)d_in[21];
  const float* eregs = (const float*)d_in[22];
  const float* gregs = (const float*)d_in[23];
  const float* W_ead = (const float*)d_in[24];
  const float* W_gad = (const float*)d_in[25];
  const float* W_emu = (const float*)d_in[26];
  const float* W_gmu = (const float*)d_in[27];
  const float* W_emd = (const float*)d_in[28];
  const float* W_gmd = (const float*)d_in[29];
  const float* W_ez1 = (const float*)d_in[30];
  const float* W_ez2 = (const float*)d_in[31];
  const float* W_gz  = (const float*)d_in[32];
  const float* W_ezd = (const float*)d_in[33];
  const float* W_gzd = (const float*)d_in[34];

  char* wsp = (char*)d_ws;
  size_t off = 0;
  auto alloc = [&](size_t bytes) -> void* {
    void* p = wsp + off;
    off = (off + bytes + 255) & ~(size_t)255;
    return p;
  };
  typedef unsigned short us;
  us* wqkv   = (us*)alloc((size_t)3072 * 1056 * 2);
  us* wb_gau = (us*)alloc((size_t)3072 * 1024 * 2);
  us* wb_ead = (us*)alloc((size_t)1024 * 1024 * 2);
  us* wb_gad = (us*)alloc((size_t)1024 * 1024 * 2);
  us* wb_emu = (us*)alloc((size_t)8192 * 1024 * 2);
  us* wb_gmu = (us*)alloc((size_t)8192 * 1024 * 2);
  us* wb_emd = (us*)alloc((size_t)1024 * 4096 * 2);
  us* wb_gmd = (us*)alloc((size_t)1024 * 4096 * 2);
  us* wb_ez1 = (us*)alloc((size_t)256 * 1056 * 2);
  us* wb_ez2 = (us*)alloc((size_t)64 * 128 * 2);
  us* wb_gz  = (us*)alloc((size_t)32 * 1024 * 2);
  us* wb_ezd = (us*)alloc((size_t)1024 * 32 * 2);
  us* wb_gzd = (us*)alloc((size_t)1024 * 32 * 2);
  float* nz   = (float*)alloc((size_t)2048 * 32 * 4);
  us*    xae  = (us*)alloc((size_t)2048 * 1056 * 2);
  us*    xag  = (us*)alloc((size_t)2048 * 1024 * 2);
  float* qkve = (float*)alloc((size_t)2048 * 3072 * 4);
  float* qkvg = (float*)alloc((size_t)2048 * 3072 * 4);
  us*    qb   = (us*)alloc((size_t)64 * 1024 * 64 * 2);
  us*    kbf  = (us*)alloc((size_t)64 * 1056 * 64 * 2);
  us*    vtb  = (us*)alloc((size_t)64 * 64 * 1056 * 2);
  us*    attn = (us*)alloc((size_t)4 * 1024 * 1024 * 2);
  float* up   = (float*)alloc((size_t)2048 * 8192 * 4);
  us*    glu  = (us*)alloc((size_t)2048 * 4096 * 2);
  float* dn   = (float*)alloc((size_t)2048 * 1024 * 4);
  float* h1   = (float*)alloc((size_t)2048 * 256 * 4);
  us*    hh   = (us*)alloc((size_t)2048 * 128 * 2);
  float* zo   = (float*)alloc((size_t)2048 * 64 * 4);
  us*    zb   = (us*)alloc((size_t)2048 * 32 * 2);
  us*    gmub = (us*)alloc((size_t)2048 * 32 * 2);

  float* out_z   = (float*)d_out;
  float* out_enc = out_z + 65536;
  float* out_gen = out_enc + 2097152;
  float* out_mu  = out_gen + 2097152;
  float* out_sig = out_mu + 65536;
  float* out_gmu = out_sig + 65536;

  auto gemm = [&](const us* Ap, int lda, const us* Bp, const float* biasp,
                  float* Cp, int M, int N, int K) {
    dim3 g((N + 127) / 128, (M + 127) / 128);
    k_gemm<<<g, THREADS, 0, stream>>>(Ap, lda, Bp, biasp, Cp, M, N, K);
  };
#define GR(n) dim3(((n) + THREADS - 1) / THREADS)
#define CVT(src, dst, n) k_f2bfv<<<GR(n), THREADS, 0, stream>>>(src, dst, n)

  // ---- weight prep: fp32 -> bf16 (QKV mask fused for encoder up-proj) ----
  k_mask_wqkv<<<GR(3072 * 1056), THREADS, 0, stream>>>(W_eau, wqkv, 3072 * 1056);
  CVT(W_gau, wb_gau, 3072 * 1024);
  CVT(W_ead, wb_ead, 1024 * 1024);
  CVT(W_gad, wb_gad, 1024 * 1024);
  CVT(W_emu, wb_emu, 8192 * 1024);
  CVT(W_gmu, wb_gmu, 8192 * 1024);
  CVT(W_emd, wb_emd, 1024 * 4096);
  CVT(W_gmd, wb_gmd, 1024 * 4096);
  CVT(W_ez1, wb_ez1, 256 * 1056);
  CVT(W_ez2, wb_ez2, 64 * 128);
  CVT(W_gz,  wb_gz,  32 * 1024);
  CVT(W_ezd, wb_ezd, 1024 * 32);
  CVT(W_gzd, wb_gzd, 1024 * 32);

  // ---- activation prep ----
  k_mask_noise<<<GR(65536), THREADS, 0, stream>>>(noise, mask, nz, 65536);
  k_ln<<<2048, THREADS, 0, stream>>>(enc_states, nullptr, nullptr, nullptr, xae, 1056);
  k_fill_noise<<<GR(65536), THREADS, 0, stream>>>(nz, xae, 65536);
  k_ln<<<2048, THREADS, 0, stream>>>(gen_states, mask, gas, gab, xag, 1024);

  // ---- QKV projections ----
  gemm(xae, 1056, wqkv,   b_eau, qkve, 2048, 3072, 1056);
  gemm(xag, 1024, wb_gau, b_gau, qkvg, 2048, 3072, 1024);

  // ---- attention ----
  k_build_q <<<GR(4 * 1024 * 1024), THREADS, 0, stream>>>(qkve, qkvg, qb, 4 * 1024 * 1024);
  k_build_kv<<<GR(4 * 1056 * 1024), THREADS, 0, stream>>>(qkve, qkvg, eregs, gregs, kbf, vtb,
                                                          4 * 1056 * 1024);
  k_attn<<<dim3(64, 8), THREADS, 0, stream>>>(qb, kbf, vtb, emask, gmask, attn);

  gemm(attn, 1024, wb_ead, nullptr, dn, 2048, 1024, 1024);
  k_res_alpha<<<GR(2097152), THREADS, 0, stream>>>(out_enc, enc_states, dn, alpha_attn, 2097152);
  gemm(attn + (size_t)2048 * 1024, 1024, wb_gad, nullptr, dn, 2048, 1024, 1024);
  k_res_filter<<<GR(2097152), THREADS, 0, stream>>>(out_gen, gen_states, dn, gaf, mask, 2097152);

  // ---- MLP (encoder) ----
  k_ln<<<2048, THREADS, 0, stream>>>(out_enc, nullptr, nullptr, nullptr, xae, 1056);
  gemm(xae, 1056, wb_emu, nullptr, up, 2048, 8192, 1024);
  k_glu<<<GR(2048 * 4096), THREADS, 0, stream>>>(up, glu, 4096, 2048 * 4096);
  gemm(glu, 4096, wb_emd, nullptr, dn, 2048, 1024, 4096);
  k_res_alpha<<<GR(2097152), THREADS, 0, stream>>>(out_enc, out_enc, dn, alpha_mlp, 2097152);

  // ---- MLP (generator) ----
  k_ln<<<2048, THREADS, 0, stream>>>(out_gen, mask, gms, gmbp, xag, 1024);
  gemm(xag, 1024, wb_gmu, nullptr, up, 2048, 8192, 1024);
  k_glu<<<GR(2048 * 4096), THREADS, 0, stream>>>(up, glu, 4096, 2048 * 4096);
  gemm(glu, 4096, wb_gmd, nullptr, dn, 2048, 1024, 4096);
  k_res_filter<<<GR(2097152), THREADS, 0, stream>>>(out_gen, out_gen, dn, gmf, mask, 2097152);

  // ---- z path (encoder) ----
  k_ln<<<2048, THREADS, 0, stream>>>(out_enc, nullptr, nullptr, nullptr, xae, 1056);
  gemm(xae, 1056, wb_ez1, nullptr, h1, 2048, 256, 1056);
  k_glu<<<GR(2048 * 128), THREADS, 0, stream>>>(h1, hh, 128, 2048 * 128);
  gemm(hh, 128, wb_ez2, nullptr, zo, 2048, 64, 128);
  k_zfin<<<GR(65536), THREADS, 0, stream>>>(zo, nz, mask, zb, out_z, out_mu, out_sig, 65536);
  gemm(zb, 32, wb_ezd, nullptr, dn, 2048, 1024, 32);
  k_res_alpha<<<GR(2097152), THREADS, 0, stream>>>(out_enc, out_enc, dn, alpha_z, 2097152);

  // ---- z path (generator) ----
  k_ln<<<2048, THREADS, 0, stream>>>(out_gen, mask, gzs, gzb, xag, 1024);
  gemm(xag, 1024, wb_gz, nullptr, zo, 2048, 32, 1024);
  k_genmu<<<GR(65536), THREADS, 0, stream>>>(zo, mask, gmub, out_gmu, 65536);
  gemm(gmub, 32, wb_gzd, nullptr, dn, 2048, 1024, 32);
  k_res_filter<<<GR(2097152), THREADS, 0, stream>>>(out_gen, out_gen, dn, gzf, mask, 2097152);
#undef CVT
#undef GR
}
